// PTGLU_58360015618523
// MI455X (gfx1250) — matrix operand order repaired and checked by reading, not compiled, not run on hardware
//
#include <hip/hip_runtime.h>
#include <hip/hip_bf16.h>

typedef __attribute__((ext_vector_type(16))) __bf16 v16bf;
typedef __attribute__((ext_vector_type(8)))  float  v8f;

#define BM   128
#define BN   128
#define BK   32
#define LDSP 40   // bf16 pitch per LDS row = 80 bytes (16B aligned, bank-spread)

union FragAB { v16bf v; uint4 q[2]; };
union FragC  { v8f   v; float f[8]; };

// ---- gfx1250 async global->LDS copy helpers (ASYNCcnt-tracked, no VGPR roundtrip)
__device__ __forceinline__ unsigned lds_addr(const void* p) {
    // generic -> AS(3) -> byte offset from LDS base
    return (unsigned)(unsigned long long)(__attribute__((address_space(3))) const char*)p;
}
__device__ __forceinline__ void async_copy_b128(unsigned lds_byte, unsigned long long gaddr) {
    asm volatile("global_load_async_to_lds_b128 %0, %1, off"
                 :: "v"(lds_byte), "v"(gaddr) : "memory");
}
__device__ __forceinline__ void wait_async0() {
#if __has_builtin(__builtin_amdgcn_s_wait_asynccnt)
    __builtin_amdgcn_s_wait_asynccnt(0);
#else
    asm volatile("s_wait_asynccnt 0" ::: "memory");
#endif
}

// ---------------------------------------------------------------------------
// Prolog 1: pack [x | h_prev] -> bf16 A1 ; h_prev -> bf16 into A2[:,2048:3072]
// ---------------------------------------------------------------------------
__global__ __launch_bounds__(256)
void pack_inputs(const float* __restrict__ x, const float* __restrict__ hprev,
                 __bf16* __restrict__ A1, __bf16* __restrict__ A2)
{
    size_t idx = (size_t)blockIdx.x * blockDim.x + threadIdx.x;   // over 4096*2048
    int r = (int)(idx >> 11);
    int c = (int)(idx & 2047);
    float v = (c < 1024) ? x[(size_t)r * 1024 + c]
                         : hprev[(size_t)r * 1024 + (c - 1024)];
    __bf16 b = (__bf16)v;
    A1[idx] = b;
    if (c >= 1024) A2[(size_t)r * 3072 + 2048 + (c - 1024)] = b;
}

// ---------------------------------------------------------------------------
// Prolog 2: LDS-tiled transpose + f32->bf16 convert of weights.
// dst[n][k] (column-major, ld=K) = src[k][col]; src = A for n<split else B.
// ---------------------------------------------------------------------------
__global__ __launch_bounds__(256)
void transpose_bf16(const float* __restrict__ srcA, const float* __restrict__ srcB,
                    int split, int ldsrc, int K, __bf16* __restrict__ dst)
{
    __shared__ float t[32][33];
    int n0 = blockIdx.x * 32, k0 = blockIdx.y * 32;
    const float* src = (n0 < split) ? srcA : srcB;
    int ncol = (n0 < split) ? n0 : (n0 - split);
    int tx = threadIdx.x, ty = threadIdx.y;          // (32, 8)
#pragma unroll
    for (int i = 0; i < 4; ++i) {
        int k = k0 + ty + i * 8;
        t[ty + i * 8][tx] = src[(size_t)k * ldsrc + ncol + tx];   // coalesced read
    }
    __syncthreads();
#pragma unroll
    for (int i = 0; i < 4; ++i) {
        int n = n0 + ty + i * 8;
        dst[(size_t)n * K + k0 + tx] = (__bf16)t[tx][ty + i * 8]; // coalesced write
    }
}

// ---------------------------------------------------------------------------
// Fused bf16 WMMA GEMM with double-buffered async-LDS staging.
// C = A[M,K] * Bt[N,K]^T ; 8 waves, WG tile 128x128, wave tile 64x32, K-step 32.
//   EPI 1: spiking front-end   (pot_next -> d_out, activated -> A2/A3)
//   EPI 2: z|r gates           (z -> f32 scratch, h_prev*r -> A3 tail, bf16)
//   EPI 3: candidate + blend   (h_next -> d_out)
// ---------------------------------------------------------------------------
template <int EPI>
__global__ __launch_bounds__(256)
void gemm_wmma(const __bf16* __restrict__ A, const __bf16* __restrict__ Bt,
               int K,
               const float* __restrict__ bias,  const float* __restrict__ bias2,
               const float* __restrict__ pot_prev, const float* __restrict__ tresh,
               const float* __restrict__ decay, const float* __restrict__ hprev,
               const float* __restrict__ zbuf_r, float* __restrict__ out_f32,
               float* __restrict__ zbuf_w, __bf16* __restrict__ actA2,
               __bf16* __restrict__ actA3)
{
    __shared__ __align__(16) __bf16 sA[2][BM * LDSP];
    __shared__ __align__(16) __bf16 sB[2][BN * LDSP];

    const int tid  = threadIdx.x;
    const int lane = tid & 31;
    const int wid  = tid >> 5;
    const int wm   = wid >> 2;     // 0..1 -> 64-row strip
    const int wn   = wid & 3;      // 0..3 -> 32-col strip
    const int hi   = lane >> 4;    // half-wave selector
    const int l16  = lane & 15;

    const int m0 = blockIdx.y * BM;
    const int n0 = blockIdx.x * BN;

    // staging coordinates: 512 16B chunks per tile, 2 per thread per tile
    const int row0 = tid >> 2,         seg0 = tid & 3;
    const int row1 = (tid + 256) >> 2, seg1 = (tid + 256) & 3;

    // precomputed LDS byte offsets (both buffers) and running global addresses
    unsigned laA0[2], laA1[2], laB0[2], laB1[2];
#pragma unroll
    for (int b = 0; b < 2; ++b) {
        laA0[b] = lds_addr(&sA[b][row0 * LDSP + seg0 * 8]);
        laA1[b] = lds_addr(&sA[b][row1 * LDSP + seg1 * 8]);
        laB0[b] = lds_addr(&sB[b][row0 * LDSP + seg0 * 8]);
        laB1[b] = lds_addr(&sB[b][row1 * LDSP + seg1 * 8]);
    }
    unsigned long long gA0 = (unsigned long long)(A  + (size_t)(m0 + row0) * K + seg0 * 8);
    unsigned long long gA1 = (unsigned long long)(A  + (size_t)(m0 + row1) * K + seg1 * 8);
    unsigned long long gB0 = (unsigned long long)(Bt + (size_t)(n0 + row0) * K + seg0 * 8);
    unsigned long long gB1 = (unsigned long long)(Bt + (size_t)(n0 + row1) * K + seg1 * 8);

    FragC acc[4][2];
#pragma unroll
    for (int i = 0; i < 4; ++i)
#pragma unroll
        for (int j = 0; j < 2; ++j)
#pragma unroll
            for (int e = 0; e < 8; ++e) acc[i][j].f[e] = 0.f;

    // prologue: async-stage K-tile 0 into buffer 0
    async_copy_b128(laA0[0], gA0); async_copy_b128(laA1[0], gA1);
    async_copy_b128(laB0[0], gB0); async_copy_b128(laB1[0], gB1);
    gA0 += BK * 2; gA1 += BK * 2; gB0 += BK * 2; gB1 += BK * 2;

    const int ksteps = K / BK;
    for (int ks = 0; ks < ksteps; ++ks) {
        const int cur = ks & 1;

        wait_async0();        // this wave's copies for buffer `cur` are in LDS
        __syncthreads();      // every wave's copies for buffer `cur` are in LDS

        // overlap: issue async DMA for the next K-tile into the other buffer
        if (ks + 1 < ksteps) {
            const int nb = cur ^ 1;
            async_copy_b128(laA0[nb], gA0); async_copy_b128(laA1[nb], gA1);
            async_copy_b128(laB0[nb], gB0); async_copy_b128(laB1[nb], gB1);
            gA0 += BK * 2; gA1 += BK * 2; gB0 += BK * 2; gB1 += BK * 2;
        }

        // ---- B fragments: lanes 0-15 hold K0..15, lanes 16-31 hold K16..31 ----
        FragAB bfr[2];
#pragma unroll
        for (int j = 0; j < 2; ++j) {
            const __bf16* p = &sB[cur][(wn * 32 + j * 16 + l16) * LDSP + hi * 16];
            bfr[j].q[0] = *(const uint4*)(p);
            bfr[j].q[1] = *(const uint4*)(p + 8);
        }
        // ---- A fragments: lanes 0-15 hold K0..15, lanes 16-31 hold K16..31 ----
#pragma unroll
        for (int i = 0; i < 4; ++i) {
            FragAB afr;
            const __bf16* p = &sA[cur][(wm * 64 + i * 16 + l16) * LDSP + hi * 16];
            afr.q[0] = *(const uint4*)(p);
            afr.q[1] = *(const uint4*)(p + 8);
#pragma unroll
            for (int j = 0; j < 2; ++j) {
                acc[i][j].v = __builtin_amdgcn_wmma_f32_16x16x32_bf16(
                    false, afr.v, false, bfr[j].v, (short)0, acc[i][j].v, false, false);
            }
        }
        __syncthreads();      // all reads of buffer `cur` done before it is re-staged
    }

    // ---- fused epilogue; C layout: lane l, VGPR v -> (m = v + 8*(l>>4), n = l&15)
#pragma unroll
    for (int i = 0; i < 4; ++i) {
#pragma unroll
        for (int j = 0; j < 2; ++j) {
#pragma unroll
            for (int vg = 0; vg < 8; ++vg) {
                const int m = m0 + wm * 64 + i * 16 + vg + 8 * hi;
                const int n = n0 + wn * 32 + j * 16 + l16;
                const float a = acc[i][j].f[vg];
                if (EPI == 1) {
                    float pt    = a + bias[n] + pot_prev[(size_t)m * 2048 + n];
                    bool  spike = (pt - tresh[n]) > 0.f;
                    float act   = spike ? pt : 0.f;
                    out_f32[(size_t)m * 2048 + n] = spike ? 0.f : pt * decay[n];
                    __bf16 ab = (__bf16)act;
                    actA2[(size_t)m * 3072 + n] = ab;
                    actA3[(size_t)m * 3072 + n] = ab;
                } else if (EPI == 2) {
                    if (n < 1024) {                       // z gate (uniform per tile)
                        float z = 1.f / (1.f + __expf(-(a + bias[n])));
                        zbuf_w[(size_t)m * 1024 + n] = z;
                    } else {                              // r gate
                        int   c = n - 1024;
                        float r = 1.f / (1.f + __expf(-(a + bias2[c])));
                        actA3[(size_t)m * 3072 + 2048 + c] =
                            (__bf16)(hprev[(size_t)m * 1024 + c] * r);
                    }
                } else {
                    float nv = tanhf(a + bias[n]);
                    float z  = zbuf_r[(size_t)m * 1024 + n];
                    float h  = hprev[(size_t)m * 1024 + n];
                    out_f32[(size_t)m * 1024 + n] = (1.f - z) * h + z * nv;
                }
            }
        }
    }
}

// ---------------------------------------------------------------------------
extern "C" void kernel_launch(void* const* d_in, const int* in_sizes, int n_in,
                              void* d_out, int out_size, void* d_ws, size_t ws_size,
                              hipStream_t stream)
{
    (void)in_sizes; (void)n_in; (void)out_size; (void)ws_size;

    const float* x        = (const float*)d_in[0];
    const float* h_prev   = (const float*)d_in[1];
    const float* pot_prev = (const float*)d_in[2];
    const float* W_in     = (const float*)d_in[3];
    const float* b_in     = (const float*)d_in[4];
    const float* tresh    = (const float*)d_in[5];
    const float* decay    = (const float*)d_in[6];
    const float* W_z      = (const float*)d_in[7];
    const float* b_z      = (const float*)d_in[8];
    const float* W_r      = (const float*)d_in[9];
    const float* b_r      = (const float*)d_in[10];
    const float* W_n      = (const float*)d_in[11];
    const float* b_n      = (const float*)d_in[12];

    // workspace layout (all offsets 256B-aligned)
    char* ws = (char*)d_ws;
    __bf16* A1   = (__bf16*)(ws + 0);            // [4096,2048] bf16  16 MB
    __bf16* A2   = (__bf16*)(ws + 16777216);     // [4096,3072] bf16  24 MB
    __bf16* A3   = (__bf16*)(ws + 41943040);     // [4096,3072] bf16  24 MB
    __bf16* WtIn = (__bf16*)(ws + 67108864);     // [2048,2048] bf16   8 MB
    __bf16* WtZR = (__bf16*)(ws + 75497472);     // [2048,3072] bf16  12 MB
    __bf16* WtN  = (__bf16*)(ws + 88080384);     // [1024,3072] bf16   6 MB
    float*  Zbuf = (float*) (ws + 94371840);     // [4096,1024] f32   16 MB

    float* outH   = (float*)d_out;                       // h_next  [4096,1024]
    float* outPot = (float*)d_out + (size_t)4096 * 1024; // pot_next[4096,2048]

    // prolog: pack activations, transpose-convert weights to column-major bf16
    pack_inputs<<<(4096 * 2048) / 256, 256, 0, stream>>>(x, h_prev, A1, A2);
    transpose_bf16<<<dim3(64, 64), dim3(32, 8), 0, stream>>>(W_in, W_in, 2048, 2048, 2048, WtIn);
    transpose_bf16<<<dim3(64, 96), dim3(32, 8), 0, stream>>>(W_z, W_r, 1024, 1024, 3072, WtZR);
    transpose_bf16<<<dim3(32, 96), dim3(32, 8), 0, stream>>>(W_n, W_n, 1024, 1024, 3072, WtN);

    // GEMM1: new_x + spiking front-end  (M=4096, N=2048, K=2048)
    gemm_wmma<1><<<dim3(16, 32), 256, 0, stream>>>(
        A1, WtIn, 2048, b_in, nullptr, pot_prev, tresh, decay,
        nullptr, nullptr, outPot, nullptr, A2, A3);

    // GEMM2: [z|r] gates fused  (M=4096, N=2048, K=3072)
    gemm_wmma<2><<<dim3(16, 32), 256, 0, stream>>>(
        A2, WtZR, 3072, b_z, b_r, nullptr, nullptr, nullptr,
        h_prev, nullptr, nullptr, Zbuf, nullptr, A3);

    // GEMM3: candidate + GRU blend  (M=4096, N=1024, K=3072)
    gemm_wmma<3><<<dim3(8, 32), 256, 0, stream>>>(
        A3, WtN, 3072, b_n, nullptr, nullptr, nullptr, nullptr,
        h_prev, Zbuf, outH, nullptr, nullptr, nullptr);
}
